// MultiRNBlock_41540923687558
// MI455X (gfx1250) — matrix operand order repaired and checked by reading, not compiled, not run on hardware
//
#include <hip/hip_runtime.h>

typedef __attribute__((ext_vector_type(16))) _Float16 v16h;
typedef __attribute__((ext_vector_type(8)))  _Float16 v8h;
typedef __attribute__((ext_vector_type(8)))  float    v8f;
typedef __attribute__((ext_vector_type(4)))  int      v4i_;
typedef __attribute__((ext_vector_type(8)))  int      v8i_;

#define BB 2
#define NN 512
#define LL 128
#define HH 256
#define CH 32          // hyb rows staged per TDM chunk (32 * 512B = 16 KB)

static __device__ inline v8h relu8(v8h x) {
  v8h z = { (_Float16)0, (_Float16)0, (_Float16)0, (_Float16)0,
            (_Float16)0, (_Float16)0, (_Float16)0, (_Float16)0 };
  return __builtin_elementwise_max(x, z);
}

// ---------------------------------------------------------------------------
// TDM: 1-D contiguous copy global -> LDS via tensor_load_to_lds (VIMAGE).
// D# group0: count=1 | lds_addr | global_addr | type=2 ("image")
// D# group1: data_size=8B, tensor_dim0 = tile_dim0 = nbytes/8 units, 1-D tile.
// Issued by one wave only; completion tracked with TENSORcnt.
// ---------------------------------------------------------------------------
static __device__ inline void tdm_load_chunk(const _Float16* gsrc, unsigned lds_off,
                                             int nbytes) {
  unsigned units = (unsigned)nbytes >> 3;              // 8-byte elements
  unsigned long long ga = (unsigned long long)(size_t)gsrc;
  v4i_ g0;
  g0[0] = 1;                                           // count=1, user descriptor
  g0[1] = (int)lds_off;                                // lds_addr (bytes)
  g0[2] = (int)(unsigned)ga;                           // global_addr[31:0]
  g0[3] = (int)((unsigned)((ga >> 32) & 0x01FFFFFFu) | (2u << 30)); // [56:32]|type=2
  v8i_ g1;
  g1[0] = (int)(3u << 16);                             // data_size = 3 (8 bytes)
  g1[1] = (int)((units & 0xFFFFu) << 16);              // tensor_dim0[15:0]
  g1[2] = (int)((units >> 16) | (1u << 16));           // tensor_dim0[31:16]|td1=1
  g1[3] = (int)(units << 16);                          // tile_dim0 = units
  g1[4] = 0;                                           // tile_dim1=0 (1-D), tile_dim2=0
  g1[5] = (int)units;                                  // tensor_dim0_stride
  g1[6] = 0;
  g1[7] = 0;
  asm volatile("tensor_load_to_lds %0, %1" :: "s"(g0), "s"(g1) : "memory");
}

// ---------------------------------------------------------------------------
// K1: hx[i,b,n,h] = (A_i @ w1[i, L:, :])[h]          (f16)
//     hyb[i,b,m,h] = (Bs_i @ w1[i, :L, :])[h] + b1   (f16)
// ---------------------------------------------------------------------------
__global__ void prep_hx_hyb(const float* __restrict__ X, const float* __restrict__ Y,
                            const float* __restrict__ w1, const float* __restrict__ b1,
                            _Float16* __restrict__ hx, _Float16* __restrict__ hyb) {
  int a   = blockIdx.x >> 10;      // 0..7  -> (i, which)
  int row = blockIdx.x & 1023;     // b*512 + n
  int i = a >> 1;
  int which = a & 1;               // 0 = hx, 1 = hyb
  const float* src;
  if (which == 0) src = (i < 2) ? X : Y;                 // A set: X,X,Y,Y
  else            src = (i == 1 || i == 3) ? Y : X;      // B set: X,Y,X,Y
  int off = (which == 0) ? LL : 0; // hx uses w1 rows [L:2L), hy uses [0:L)

  __shared__ float rowbuf[LL];
  if (threadIdx.x < LL) rowbuf[threadIdx.x] = src[row * LL + threadIdx.x];
  __syncthreads();

  int h = threadIdx.x;             // 0..255
  float acc = 0.f;
  #pragma unroll 4
  for (int k = 0; k < LL; ++k)
    acc += rowbuf[k] * w1[(i * (2 * LL) + off + k) * HH + h];
  if (which) acc += b1[i * HH + h];
  _Float16* dst = which ? hyb : hx;
  dst[(i * (BB * NN) + row) * HH + h] = (_Float16)acc;
}

// ---------------------------------------------------------------------------
// K2: repack w2 (f32 [4][256][128]) into WMMA B-fragment lane order (f16):
//     frag(i, lt, ks): lane l holds col = lt*16+(l&15),
//                      k = ks*32 + (l>>4)*16 + e, e = 0..15 contiguous.
// ---------------------------------------------------------------------------
__global__ void pack_w2(const float* __restrict__ w2, _Float16* __restrict__ w2f) {
  int blk = blockIdx.x;            // 4*8*8 = 256
  int i = blk >> 6, lt = (blk >> 3) & 7, ks = blk & 7;
  int lane = threadIdx.x;          // 0..31
  int col = lt * 16 + (lane & 15);
  int kb  = ks * 32 + ((lane >> 4) << 4);
  _Float16* dst = &w2f[(((i * 64) + lt * 8 + ks) * 32 + lane) * 16];
  #pragma unroll
  for (int e = 0; e < 16; ++e)
    dst[e] = (_Float16)w2[(i * HH + kb + e) * LL + col];
}

// ---------------------------------------------------------------------------
// K3: fused relu-GEMM-maxpool via v_wmma_f32_16x16x32_f16.
// grid: 256 = (block i)*64 + b*32 + n-tile. 8 waves; wave w owns l-tile w.
// hyb rows staged to LDS in double-buffered 32-row chunks by the Tensor Data
// Mover (tensor_load_to_lds + s_wait_tensorcnt), overlapping DMA with WMMA.
// ---------------------------------------------------------------------------
__global__ void __launch_bounds__(256)
rn_main(const float* __restrict__ X, const float* __restrict__ Y,
        const _Float16* __restrict__ hx, const _Float16* __restrict__ hyb,
        const _Float16* __restrict__ w2f, const float* __restrict__ b2,
        float* __restrict__ z1) {
  int wg = blockIdx.x;
  int i  = wg >> 6;
  int b  = (wg >> 5) & 1;
  int nt = wg & 31;
  int lane = threadIdx.x & 31;
  int lt   = threadIdx.x >> 5;     // wave id == l-tile (0..7)
  bool wave0 = (threadIdx.x < 32);

  __shared__ __align__(16) _Float16 shx[16 * HH];        // 8 KB hx tile
  __shared__ __align__(16) _Float16 hybbuf[2][CH * HH];  // 2 x 16 KB hyb chunks
  {
    const uint32_t* srcw = (const uint32_t*)&hx[((i * BB + b) * NN + nt * 16) * HH];
    uint32_t* dstw = (uint32_t*)shx;
    for (int idx = threadIdx.x; idx < (16 * HH) / 2; idx += 256)
      dstw[idx] = srcw[idx];
  }

  const _Float16* hybbase = &hyb[((i * BB + b) * NN) * HH];
  unsigned lds_buf[2];
  lds_buf[0] = (unsigned)(size_t)&hybbuf[0][0];   // flat addr low 32b == LDS offset
  lds_buf[1] = (unsigned)(size_t)&hybbuf[1][0];

  if (wave0)                                       // prefetch chunk 0 -> buf 0
    tdm_load_chunk(hybbase, lds_buf[0], CH * HH * 2);

  // B fragments (w2) — register-resident across the whole m loop: 8 x v16h.
  v16h bf[8];
  #pragma unroll
  for (int ks = 0; ks < 8; ++ks) {
    const v8h* p = (const v8h*)&w2f[(((i * 64) + lt * 8 + ks) * 32 + lane) * 16];
    v8h lo = p[0], hi = p[1];
    bf[ks] = __builtin_shufflevector(lo, hi, 0,1,2,3,4,5,6,7,8,9,10,11,12,13,14,15);
  }

  int r    = lane & 15;            // A-matrix row (n within tile)
  int ksub = (lane >> 4) << 4;     // K sub-offset per lane half

  v8f pooled;
  #pragma unroll
  for (int j = 0; j < 8; ++j) pooled[j] = -3.0e38f;

  for (int c = 0; c < NN / CH; ++c) {
    if (wave0) __builtin_amdgcn_s_wait_tensorcnt(0);   // chunk c landed in LDS
    __syncthreads();                                   // visible to all waves;
                                                       // prev chunk fully consumed
    if (wave0 && (c + 1) < NN / CH)                    // async-prefetch chunk c+1
      tdm_load_chunk(hybbase + (size_t)(c + 1) * CH * HH,
                     lds_buf[(c + 1) & 1], CH * HH * 2);

    const _Float16* chunk = &hybbuf[c & 1][0];
    for (int mm = 0; mm < CH; ++mm) {
      const _Float16* hrow = &chunk[mm * HH];

      v8f cacc;
      #pragma unroll
      for (int j = 0; j < 8; ++j) cacc[j] = 0.f;

      #pragma unroll
      for (int ks = 0; ks < 8; ++ks) {
        int kb1 = ks * 32 + ksub;    // lane's two contiguous 8-half K runs
        int kb2 = kb1 + 8;
        v8h x0 = *(const v8h*)&shx[r * HH + kb1];
        v8h x1 = *(const v8h*)&shx[r * HH + kb2];
        v8h y0 = *(const v8h*)&hrow[kb1];
        v8h y1 = *(const v8h*)&hrow[kb2];
        v8h s0 = relu8(x0 + y0);     // v_pk_add_f16 + v_pk_max_num_f16
        v8h s1 = relu8(x1 + y1);
        v16h amat = __builtin_shufflevector(s0, s1,
                        0,1,2,3,4,5,6,7,8,9,10,11,12,13,14,15);
        cacc = __builtin_amdgcn_wmma_f32_16x16x32_f16(
                   false, amat, false, bf[ks], (short)0, cacc, false, false);
      }
      pooled = __builtin_elementwise_max(pooled, cacc);  // max-pool over m
    }
  }

  // Z1 = A_in + pooled + b2   (C/D layout: VGPR j -> row j (lanes<16) / j+8)
  const float* base = (i < 2) ? X : Y;
  int col = lane & 15;
  int lg  = lt * 16 + col;
  int rhi = (lane >> 4) << 3;
  #pragma unroll
  for (int j = 0; j < 8; ++j) {
    int n = nt * 16 + rhi + j;
    float v = base[(b * NN + n) * LL + lg] + pooled[j] + b2[i * LL + lg];
    z1[((i * BB + b) * NN + n) * LL + lg] = v;
  }
}

// ---------------------------------------------------------------------------
// K4: per-block FFN  z2 = z1 + relu(z1@fw1+fb1)@fw2 + fb2
// ---------------------------------------------------------------------------
__global__ void ffn_block(const float* __restrict__ z1,
                          const float* __restrict__ fw1, const float* __restrict__ fb1,
                          const float* __restrict__ fw2, const float* __restrict__ fb2,
                          float* __restrict__ z2) {
  int i   = blockIdx.x >> 10;
  int row = blockIdx.x & 1023;
  __shared__ float zr[LL];
  __shared__ float u[HH];
  int t = threadIdx.x;             // 0..255
  if (t < LL) zr[t] = z1[(i * (BB * NN) + row) * LL + t];
  __syncthreads();
  float acc = fb1[i * HH + t];
  #pragma unroll 4
  for (int k = 0; k < LL; ++k) acc += zr[k] * fw1[(i * LL + k) * HH + t];
  u[t] = acc > 0.f ? acc : 0.f;
  __syncthreads();
  if (t < LL) {
    float a2 = fb2[i * LL + t];
    #pragma unroll 4
    for (int k = 0; k < HH; ++k) a2 += u[k] * fw2[(i * HH + k) * LL + t];
    z2[(i * (BB * NN) + row) * LL + t] = zr[t] + a2;
  }
}

// ---------------------------------------------------------------------------
// K5: X_out = X + relu([Z_XX|Z_XY]@fcX_w + fcX_b);  Y_out analog ([Z_YY|Z_YX])
// ---------------------------------------------------------------------------
__global__ void final_combine(const float* __restrict__ X, const float* __restrict__ Y,
                              const float* __restrict__ z2,
                              const float* __restrict__ fcXw, const float* __restrict__ fcXb,
                              const float* __restrict__ fcYw, const float* __restrict__ fcYb,
                              float* __restrict__ out) {
  int sel = blockIdx.x >> 10;      // 0 -> X_out, 1 -> Y_out
  int row = blockIdx.x & 1023;
  __shared__ float zc[2 * LL];
  int t = threadIdx.x;             // 0..127
  int iA = sel ? 3 : 0;            // Z_YY or Z_XX
  int iB = sel ? 2 : 1;            // Z_YX or Z_XY
  zc[t]      = z2[(iA * (BB * NN) + row) * LL + t];
  zc[LL + t] = z2[(iB * (BB * NN) + row) * LL + t];
  __syncthreads();
  const float* W    = sel ? fcYw : fcXw;
  const float* bias = sel ? fcYb : fcXb;
  const float* base = sel ? Y : X;
  float acc = bias[t];
  #pragma unroll 4
  for (int k = 0; k < 2 * LL; ++k) acc += zc[k] * W[k * LL + t];
  acc = acc > 0.f ? acc : 0.f;
  out[sel * (BB * NN * LL) + row * LL + t] = base[row * LL + t] + acc;
}

// ---------------------------------------------------------------------------
extern "C" void kernel_launch(void* const* d_in, const int* in_sizes, int n_in,
                              void* d_out, int out_size, void* d_ws, size_t ws_size,
                              hipStream_t stream) {
  const float* X     = (const float*)d_in[0];
  const float* Y     = (const float*)d_in[1];
  const float* rn_w1 = (const float*)d_in[2];
  const float* rn_b1 = (const float*)d_in[3];
  const float* rn_w2 = (const float*)d_in[4];
  const float* rn_b2 = (const float*)d_in[5];
  const float* fc_w1 = (const float*)d_in[6];
  const float* fc_b1 = (const float*)d_in[7];
  const float* fc_w2 = (const float*)d_in[8];
  const float* fc_b2 = (const float*)d_in[9];
  const float* fcX_w = (const float*)d_in[10];
  const float* fcX_b = (const float*)d_in[11];
  const float* fcY_w = (const float*)d_in[12];
  const float* fcY_b = (const float*)d_in[13];
  float* out = (float*)d_out;

  char* ws = (char*)d_ws;
  const size_t hx_bytes  = (size_t)4 * BB * NN * HH * 2;   // 2 MB f16
  const size_t w2f_bytes = (size_t)4 * 64 * 32 * 16 * 2;   // 256 KB f16
  _Float16* hx  = (_Float16*)(ws);
  _Float16* hyb = (_Float16*)(ws + hx_bytes);
  _Float16* w2f = (_Float16*)(ws + 2 * hx_bytes);
  float*    z1  = (float*)(ws + 2 * hx_bytes + w2f_bytes);
  float*    z2  = z1 + (size_t)4 * BB * NN * LL;

  prep_hx_hyb<<<dim3(8 * BB * NN), dim3(HH), 0, stream>>>(X, Y, rn_w1, rn_b1, hx, hyb);
  pack_w2<<<dim3(256), dim3(32), 0, stream>>>(rn_w2, w2f);
  rn_main<<<dim3(256), dim3(256), 0, stream>>>(X, Y, hx, hyb, w2f, rn_b2, z1);
  ffn_block<<<dim3(4 * BB * NN), dim3(HH), 0, stream>>>(z1, fc_w1, fc_b1, fc_w2, fc_b2, z2);
  final_combine<<<dim3(2 * BB * NN), dim3(LL), 0, stream>>>(X, Y, z2, fcX_w, fcX_b,
                                                            fcY_w, fcY_b, out);
}
